// GAT_29850022707974
// MI455X (gfx1250) — matrix operand order repaired and checked by reading, not compiled, not run on hardware
//
#include <hip/hip_runtime.h>
#include <math.h>

// ---------------------------------------------------------------------------
// GAT (2-layer, PyG-style) for gfx1250 / MI455X.
// GEMMs via v_wmma_f32_16x16x32_f16 (wave32, one wave per 16x16 tile).
// Attention: segment max (atomicMax w/ monotone uint map) -> exp/segment sum
// -> alpha normalize -> float4-vectorized weighted scatter-add
// (global_load_b128 + global_atomic_add_f32).
// ---------------------------------------------------------------------------

typedef __attribute__((ext_vector_type(16))) _Float16 v16h;
typedef __attribute__((ext_vector_type(8)))  _Float16 v8h;
typedef __attribute__((ext_vector_type(8)))  float    v8f;

#define N_NODES   50000
#define N_EDGES   800000
#define E_TOT     (N_EDGES + N_NODES)   // self-loops appended
#define IN_DIM    128
#define HEADS     4
#define HID       32
#define F1        (HEADS * HID)         // 128
#define OUT_DIM   64
#define NEG_SLOPE 0.2f

// ---- helpers ---------------------------------------------------------------

__device__ __forceinline__ unsigned f2u_mono(float f) {
  unsigned u = __float_as_uint(f);
  return (u & 0x80000000u) ? ~u : (u | 0x80000000u);
}
__device__ __forceinline__ float u2f_mono(unsigned u) {
  unsigned r = (u & 0x80000000u) ? (u & 0x7fffffffu) : ~u;
  return __uint_as_float(r);
}
__device__ __forceinline__ void edge_sd(const int* __restrict__ ei, int e,
                                        int& s, int& d) {
  if (e < N_EDGES) { s = ei[e]; d = ei[N_EDGES + e]; }
  else             { s = d = e - N_EDGES; }          // self-loop
}
__device__ __forceinline__ int edge_dst(const int* __restrict__ ei, int e) {
  return (e < N_EDGES) ? ei[N_EDGES + e] : (e - N_EDGES);
}

// ---- small utility kernels -------------------------------------------------

__global__ void k_cast16(const float* __restrict__ src,
                         _Float16* __restrict__ dst, int n) {
  int i = blockIdx.x * blockDim.x + threadIdx.x;
  if (i < n) dst[i] = (_Float16)src[i];
}

// W[K][N] (row-major) -> Wt[N][K] f16
__global__ void k_transpose16(const float* __restrict__ W,
                              _Float16* __restrict__ Wt, int K, int N) {
  int i = blockIdx.x * blockDim.x + threadIdx.x;
  if (i >= K * N) return;
  int k = i / N, n = i % N;
  Wt[n * K + k] = (_Float16)W[k * N + n];
}

__global__ void k_fill_u32(unsigned* __restrict__ p, unsigned v, int n) {
  int i = blockIdx.x * blockDim.x + threadIdx.x;
  if (i < n) p[i] = v;
}
__global__ void k_fill_f32(float* __restrict__ p, float v, int n) {
  int i = blockIdx.x * blockDim.x + threadIdx.x;
  if (i < n) p[i] = v;
}
__global__ void k_init_bias(float* __restrict__ out,
                            const float* __restrict__ b, int F, int n) {
  int i = blockIdx.x * blockDim.x + threadIdx.x;
  if (i < n) out[i] = b[i % F];
}

// ---- WMMA GEMM: D[M][N] = A[M][K] * Bt[N][K]^T, f16 in, f32 accum ----------
// One wave per 16x16 tile. blockDim.x == 32 (full EXEC, as WMMA requires).

__global__ void k_wmma_gemm(const _Float16* __restrict__ A,
                            const _Float16* __restrict__ Bt,
                            float* __restrict__ D,
                            int N, int K) {
  const int mt   = blockIdx.x;           // M tile
  const int nt   = blockIdx.y;           // N tile
  const int lane = threadIdx.x & 31;
  const int half = lane >> 4;
  const int r    = lane & 15;

  const _Float16* arow = A  + (size_t)(mt * 16 + r) * K;
  const _Float16* brow = Bt + (size_t)(nt * 16 + r) * K;

  v8f c = {};
#pragma unroll
  for (int k0 = 0; k0 < 128; k0 += 32) {
    union { v16h v; v8h h[2]; } a;
    a.h[0] = *(const v8h*)(arow + k0 + half * 8);        // K = k0+half*8 .. +7
    a.h[1] = *(const v8h*)(arow + k0 + 16 + half * 8);   // K = k0+16+half*8 ..
    union { v16h v; v8h h[2]; } bb;
    bb.h[0] = *(const v8h*)(brow + k0 + half * 8);       // K = k0+half*8 .. +7
    bb.h[1] = *(const v8h*)(brow + k0 + 16 + half * 8);  // K = k0+16+half*8 ..
    v16h b = bb.v;
    c = __builtin_amdgcn_wmma_f32_16x16x32_f16(
        /*neg_a=*/false, a.v, /*neg_b=*/false, b,
        /*c_mod=*/(short)0, c, /*reuse_a=*/false, /*reuse_b=*/false);
  }

  const int col = nt * 16 + r;
#pragma unroll
  for (int v = 0; v < 8; ++v)
    D[(size_t)(mt * 16 + v + half * 8) * N + col] = c[v];
}

// ---- per-node attention logits: alpha_s/d[n,h] = <h[n,h,:], att[h,:]> ------

__global__ void k_node_alpha(const float* __restrict__ h,
                             const float* __restrict__ att_s,
                             const float* __restrict__ att_d,
                             float* __restrict__ as_, float* __restrict__ ad_,
                             int H, int C) {
  int i = blockIdx.x * blockDim.x + threadIdx.x;
  if (i >= N_NODES * H) return;
  int n = i / H, hh = i % H;
  const float* row = h + (size_t)n * H * C + (size_t)hh * C;
  const float* vs  = att_s + hh * C;
  const float* vd  = att_d + hh * C;
  float s = 0.f, d = 0.f;
  for (int c = 0; c < C; ++c) { float v = row[c]; s += v * vs[c]; d += v * vd[c]; }
  as_[i] = s; ad_[i] = d;
}

// ---- edge pass 1: leaky-relu logit, segment max into mmax[dst,h] -----------

__global__ void k_edge_logit_max(const int* __restrict__ ei,
                                 const float* __restrict__ as_,
                                 const float* __restrict__ ad_,
                                 float* __restrict__ elog,
                                 unsigned* __restrict__ mmax, int H) {
  int i = blockIdx.x * blockDim.x + threadIdx.x;
  if (i >= E_TOT * H) return;
  int e = i / H, hh = i % H;
  int s, d; edge_sd(ei, e, s, d);
  float x = as_[s * H + hh] + ad_[d * H + hh];
  x = (x > 0.f) ? x : NEG_SLOPE * x;
  elog[i] = x;
  atomicMax(&mmax[d * H + hh], f2u_mono(x));
}

// ---- edge pass 2: ex = exp(l - m[dst]); denom[dst,h] += ex -----------------

__global__ void k_edge_expsum(const int* __restrict__ ei,
                              float* __restrict__ elog,
                              const unsigned* __restrict__ mmax,
                              float* __restrict__ denom, int H) {
  int i = blockIdx.x * blockDim.x + threadIdx.x;
  if (i >= E_TOT * H) return;
  int e = i / H, hh = i % H;
  int d = edge_dst(ei, e);
  float ex = expf(elog[i] - u2f_mono(mmax[d * H + hh]));
  elog[i] = ex;                                   // overwrite with exp value
  atomicAdd(&denom[d * H + hh], ex);
}

// ---- edge pass 2.5: normalize in place: alpha = ex / denom[dst,h] ----------

__global__ void k_edge_alpha(const int* __restrict__ ei,
                             float* __restrict__ elog,
                             const float* __restrict__ denom, int H) {
  int i = blockIdx.x * blockDim.x + threadIdx.x;
  if (i >= E_TOT * H) return;
  int e = i / H, hh = i % H;
  int d = edge_dst(ei, e);
  elog[i] = elog[i] / denom[d * H + hh];
}

// ---- edge pass 3: agg[dst, f..f+3] += h[src, f..f+3] * alpha[e, h] ---------
// one thread per (edge, group of 4 features): b128 gather + 4 f32 atomics

__global__ void k_edge_scatter4(const int* __restrict__ ei,
                                const float* __restrict__ alpha,
                                const float* __restrict__ hfeat,
                                float* __restrict__ agg, int H, int C) {
  const int F  = H * C;
  const int G  = F >> 2;                         // float4 groups per edge
  long long i = (long long)blockIdx.x * blockDim.x + threadIdx.x;
  if (i >= (long long)E_TOT * G) return;
  int e = (int)(i / G), g = (int)(i % G);
  int f = g << 2;
  int hh = f / C;
  int s, d; edge_sd(ei, e, s, d);
  float a = alpha[(size_t)e * H + hh];
  const float4 v = *(const float4*)(hfeat + (size_t)s * F + f);
  float* o = agg + (size_t)d * F + f;
  atomicAdd(o + 0, v.x * a);
  atomicAdd(o + 1, v.y * a);
  atomicAdd(o + 2, v.z * a);
  atomicAdd(o + 3, v.w * a);
}

// ---- bias + relu + cast to f16 (layer-1 output -> layer-2 GEMM input) ------

__global__ void k_bias_relu_cast(const float* __restrict__ agg,
                                 const float* __restrict__ b,
                                 _Float16* __restrict__ out16, int F, int n) {
  int i = blockIdx.x * blockDim.x + threadIdx.x;
  if (i >= n) return;
  float v = agg[i] + b[i % F];
  out16[i] = (_Float16)(v > 0.f ? v : 0.f);
}

// ---------------------------------------------------------------------------

static inline int cdiv(long long a, long long b) { return (int)((a + b - 1) / b); }

extern "C" void kernel_launch(void* const* d_in, const int* in_sizes, int n_in,
                              void* d_out, int out_size, void* d_ws, size_t ws_size,
                              hipStream_t stream) {
  const float* x    = (const float*)d_in[0];
  const int*   ei   = (const int*)d_in[1];   // [2, N_EDGES] (src row, dst row)
  const float* W1   = (const float*)d_in[2]; // [128,128]
  const float* as1w = (const float*)d_in[3]; // [4,32]
  const float* ad1w = (const float*)d_in[4];
  const float* b1   = (const float*)d_in[5]; // [128]
  const float* W2   = (const float*)d_in[6]; // [128,64]
  const float* as2w = (const float*)d_in[7]; // [1,64]
  const float* ad2w = (const float*)d_in[8];
  const float* b2   = (const float*)d_in[9]; // [64]
  float* out = (float*)d_out;                // [N_NODES, 64]

  // bump allocator in workspace (256B aligned)
  char* wp = (char*)d_ws;
  auto alloc = [&](size_t bytes) -> void* {
    void* p = (void*)wp;
    wp += (bytes + 255) & ~(size_t)255;
    return p;
  };

  _Float16* x16    = (_Float16*)alloc((size_t)N_NODES * IN_DIM * 2);
  _Float16* W1t    = (_Float16*)alloc((size_t)F1 * IN_DIM * 2);       // [N=128][K=128]
  _Float16* W2t    = (_Float16*)alloc((size_t)OUT_DIM * F1 * 2);      // [N=64][K=128]
  float*    h1     = (float*)   alloc((size_t)N_NODES * F1 * 4);
  float*    as1    = (float*)   alloc((size_t)N_NODES * HEADS * 4);
  float*    ad1    = (float*)   alloc((size_t)N_NODES * HEADS * 4);
  unsigned* m1     = (unsigned*)alloc((size_t)N_NODES * HEADS * 4);
  float*    den1   = (float*)   alloc((size_t)N_NODES * HEADS * 4);
  float*    elog1  = (float*)   alloc((size_t)E_TOT * HEADS * 4);
  float*    agg1   = (float*)   alloc((size_t)N_NODES * F1 * 4);
  _Float16* h1a16  = (_Float16*)alloc((size_t)N_NODES * F1 * 2);
  float*    h2     = (float*)   alloc((size_t)N_NODES * OUT_DIM * 4);
  float*    as2    = (float*)   alloc((size_t)N_NODES * 4);
  float*    ad2    = (float*)   alloc((size_t)N_NODES * 4);
  unsigned* m2     = (unsigned*)alloc((size_t)N_NODES * 4);
  float*    den2   = (float*)   alloc((size_t)N_NODES * 4);
  float*    elog2  = (float*)   alloc((size_t)E_TOT * 4);
  (void)ws_size; (void)in_sizes; (void)n_in; (void)out_size;

  const int B = 256;
  const unsigned NEG_INF_U = 0x007FFFFFu;  // f2u_mono(-inf)

  // ---------------- layer 1 ----------------
  k_cast16<<<cdiv((long long)N_NODES * IN_DIM, B), B, 0, stream>>>(x, x16, N_NODES * IN_DIM);
  k_transpose16<<<cdiv(IN_DIM * F1, B), B, 0, stream>>>(W1, W1t, IN_DIM, F1);
  k_transpose16<<<cdiv(F1 * OUT_DIM, B), B, 0, stream>>>(W2, W2t, F1, OUT_DIM);

  { dim3 g(N_NODES / 16, F1 / 16);
    k_wmma_gemm<<<g, 32, 0, stream>>>(x16, W1t, h1, F1, IN_DIM); }

  k_node_alpha<<<cdiv(N_NODES * HEADS, B), B, 0, stream>>>(h1, as1w, ad1w, as1, ad1, HEADS, HID);

  k_fill_u32<<<cdiv(N_NODES * HEADS, B), B, 0, stream>>>(m1, NEG_INF_U, N_NODES * HEADS);
  k_fill_f32<<<cdiv(N_NODES * HEADS, B), B, 0, stream>>>(den1, 0.f, N_NODES * HEADS);
  k_fill_f32<<<cdiv((long long)N_NODES * F1, B), B, 0, stream>>>(agg1, 0.f, N_NODES * F1);

  k_edge_logit_max<<<cdiv((long long)E_TOT * HEADS, B), B, 0, stream>>>(ei, as1, ad1, elog1, m1, HEADS);
  k_edge_expsum   <<<cdiv((long long)E_TOT * HEADS, B), B, 0, stream>>>(ei, elog1, m1, den1, HEADS);
  k_edge_alpha    <<<cdiv((long long)E_TOT * HEADS, B), B, 0, stream>>>(ei, elog1, den1, HEADS);
  k_edge_scatter4 <<<cdiv((long long)E_TOT * (F1 / 4), B), B, 0, stream>>>(ei, elog1, h1, agg1, HEADS, HID);

  k_bias_relu_cast<<<cdiv((long long)N_NODES * F1, B), B, 0, stream>>>(agg1, b1, h1a16, F1, N_NODES * F1);

  // ---------------- layer 2 ----------------
  { dim3 g(N_NODES / 16, OUT_DIM / 16);
    k_wmma_gemm<<<g, 32, 0, stream>>>(h1a16, W2t, h2, OUT_DIM, F1); }

  k_node_alpha<<<cdiv(N_NODES, B), B, 0, stream>>>(h2, as2w, ad2w, as2, ad2, 1, OUT_DIM);

  k_fill_u32<<<cdiv(N_NODES, B), B, 0, stream>>>(m2, NEG_INF_U, N_NODES);
  k_fill_f32<<<cdiv(N_NODES, B), B, 0, stream>>>(den2, 0.f, N_NODES);
  k_init_bias<<<cdiv((long long)N_NODES * OUT_DIM, B), B, 0, stream>>>(out, b2, OUT_DIM, N_NODES * OUT_DIM);

  k_edge_logit_max<<<cdiv((long long)E_TOT, B), B, 0, stream>>>(ei, as2, ad2, elog2, m2, 1);
  k_edge_expsum   <<<cdiv((long long)E_TOT, B), B, 0, stream>>>(ei, elog2, m2, den2, 1);
  k_edge_alpha    <<<cdiv((long long)E_TOT, B), B, 0, stream>>>(ei, elog2, den2, 1);
  k_edge_scatter4 <<<cdiv((long long)E_TOT * (OUT_DIM / 4), B), B, 0, stream>>>(ei, elog2, h2, out, 1, OUT_DIM);
}
